// Attention_72438918414643
// MI455X (gfx1250) — matrix operand order repaired and checked by reading, not compiled, not run on hardware
//
#include <hip/hip_runtime.h>
#include <hip/hip_bf16.h>

// ---------------------------------------------------------------------------
// Attention block on gfx1250 (MI455X).  B=4, S=1024, C=2048, H=16, D=128.
//  - Dense GEMMs: bf16x3 split (hi/lo) on V_WMMA_F32_16X16X32_BF16
//  - Attention core: V_WMMA_F32_16X16X4_F32 (exact fp32 products)
//  - k/v tile staging: GLOBAL_LOAD_ASYNC_TO_LDS_B128 (ASYNCcnt path)
// ---------------------------------------------------------------------------

typedef float  v2f  __attribute__((ext_vector_type(2)));
typedef float  v8f  __attribute__((ext_vector_type(8)));
typedef __bf16 bf16_t;
typedef __bf16 v16bf __attribute__((ext_vector_type(16)));
typedef __bf16 v8bf  __attribute__((ext_vector_type(8)));
typedef __bf16 v4bf  __attribute__((ext_vector_type(4)));

// Async-to-LDS builtin parameter types (from hipcc diagnostic):
//   arg0: int __vector(4) __device__(AS1) *   arg1: AS3 counterpart
typedef int v4i_gcc __attribute__((vector_size(16)));
typedef __attribute__((address_space(1))) v4i_gcc gv4i;
typedef __attribute__((address_space(3))) v4i_gcc lv4i;

#if defined(__has_builtin)
#if __has_builtin(__builtin_amdgcn_global_load_async_to_lds_b128)
#define USE_ASYNC_LDS 1
#endif
#endif
#ifndef USE_ASYNC_LDS
#define USE_ASYNC_LDS 0
#endif

__device__ __forceinline__ v8f wmma_f32(v2f a, v2f b, v8f c) {
  return __builtin_amdgcn_wmma_f32_16x16x4_f32(false, a, false, b, (short)0, c,
                                               false, false);
}
__device__ __forceinline__ v8f wmma_bf16(v16bf a, v16bf b, v8f c) {
  return __builtin_amdgcn_wmma_f32_16x16x32_bf16(false, a, false, b, (short)0,
                                                 c, false, false);
}

struct bfpair { bf16_t h, l; };
__device__ __forceinline__ bfpair split_bf16(float x) {
  bfpair p;
  p.h = (bf16_t)x;
  p.l = (bf16_t)(x - (float)p.h);
  return p;
}

// ---------------------------------------------------------------------------
// Kernel 1/4: GEMM Out[M,N] = A[M,K] @ W[K,N] (+bias), bf16x3 split.
// BM=64, BN=128, BK=32; 256 threads = 8 waves (2x4); wave tile 32x32.
// ---------------------------------------------------------------------------
#define AST 40
#define BST 40

__global__ __launch_bounds__(256) void gemm_bf16x3_kernel(
    const float* __restrict__ A, const float* __restrict__ W,
    const float* __restrict__ bias, float* __restrict__ Out,
    int M, int N, int K) {
  __shared__ bf16_t Ah[64 * AST], Al[64 * AST];    // 5 KB each
  __shared__ bf16_t Bh[128 * BST], Bl[128 * BST];  // 10 KB each

  const int tid  = threadIdx.x;
  const int wave = tid >> 5, lane = tid & 31;
  const int half = lane >> 4, m = lane & 15;
  const int wm = wave >> 2, wn = wave & 3;
  const int bm = blockIdx.y * 64;
  const int bn = blockIdx.x * 128;
  const int sn = tid & 127;  // B staging: output column
  const int sk = tid >> 7;   // B staging: k sub-group (0..1)
  (void)M;

  v8f acc[2][2];
#pragma unroll
  for (int i = 0; i < 2; ++i)
#pragma unroll
    for (int j = 0; j < 2; ++j)
#pragma unroll
      for (int r = 0; r < 8; ++r) acc[i][j][r] = 0.0f;

  for (int kt = 0; kt < K; kt += 32) {
    __syncthreads();
    // ---- stage A tile 64x32 (row major), split to hi/lo bf16 ----
#pragma unroll
    for (int i = 0; i < 2; ++i) {
      int slot = tid + i * 256;  // 512 float4 slots
      int r = slot >> 3, c4 = slot & 7;
      float4 v = *(const float4*)(A + (size_t)(bm + r) * K + kt + c4 * 4);
      v4bf hv, lv;
      bfpair p;
      p = split_bf16(v.x); hv[0] = p.h; lv[0] = p.l;
      p = split_bf16(v.y); hv[1] = p.h; lv[1] = p.l;
      p = split_bf16(v.z); hv[2] = p.h; lv[2] = p.l;
      p = split_bf16(v.w); hv[3] = p.h; lv[3] = p.l;
      *(v4bf*)(Ah + r * AST + c4 * 4) = hv;
      *(v4bf*)(Al + r * AST + c4 * 4) = lv;
    }
    // ---- stage B tile 32x128, K-transposed to [n][k], split hi/lo ----
#pragma unroll
    for (int i = 0; i < 4; ++i) {
      const int k0 = i * 8 + sk * 4;
      v4bf hv, lv;
#pragma unroll
      for (int j = 0; j < 4; ++j) {
        float x = W[(size_t)(kt + k0 + j) * N + bn + sn];
        bfpair p = split_bf16(x);
        hv[j] = p.h;
        lv[j] = p.l;
      }
      *(v4bf*)(Bh + sn * BST + k0) = hv;
      *(v4bf*)(Bl + sn * BST + k0) = lv;
    }
    __syncthreads();

    // ---- fragments ----
    v16bf ah[2], al2[2], bh2[2], bl2[2];
#pragma unroll
    for (int i = 0; i < 2; ++i) {
      const int row = (wm * 32 + i * 16 + m) * AST;
      v8bf h0 = *(const v8bf*)(Ah + row + 8 * half);
      v8bf h1 = *(const v8bf*)(Ah + row + 16 + 8 * half);
      ah[i] = __builtin_shufflevector(h0, h1, 0, 1, 2, 3, 4, 5, 6, 7, 8, 9, 10,
                                      11, 12, 13, 14, 15);
      v8bf l0 = *(const v8bf*)(Al + row + 8 * half);
      v8bf l1 = *(const v8bf*)(Al + row + 16 + 8 * half);
      al2[i] = __builtin_shufflevector(l0, l1, 0, 1, 2, 3, 4, 5, 6, 7, 8, 9, 10,
                                       11, 12, 13, 14, 15);
    }
#pragma unroll
    for (int j = 0; j < 2; ++j) {
      const int col = (wn * 32 + j * 16 + m) * BST + 8 * half;
      v8bf h0 = *(const v8bf*)(Bh + col);
      v8bf h1 = *(const v8bf*)(Bh + col + 16);
      bh2[j] = __builtin_shufflevector(h0, h1, 0, 1, 2, 3, 4, 5, 6, 7, 8, 9, 10,
                                       11, 12, 13, 14, 15);
      v8bf l0 = *(const v8bf*)(Bl + col);
      v8bf l1 = *(const v8bf*)(Bl + col + 16);
      bl2[j] = __builtin_shufflevector(l0, l1, 0, 1, 2, 3, 4, 5, 6, 7, 8, 9, 10,
                                       11, 12, 13, 14, 15);
    }
    // ---- 3-term split product: Al*Bh + Ah*Bl + Ah*Bh ----
#pragma unroll
    for (int i = 0; i < 2; ++i)
#pragma unroll
      for (int j = 0; j < 2; ++j) {
        acc[i][j] = wmma_bf16(al2[i], bh2[j], acc[i][j]);
        acc[i][j] = wmma_bf16(ah[i], bl2[j], acc[i][j]);
        acc[i][j] = wmma_bf16(ah[i], bh2[j], acc[i][j]);
      }
  }

#pragma unroll
  for (int i = 0; i < 2; ++i)
#pragma unroll
    for (int j = 0; j < 2; ++j)
#pragma unroll
      for (int r = 0; r < 8; ++r) {
        const int row = bm + wm * 32 + i * 16 + r + 8 * half;
        const int col = bn + wn * 32 + j * 16 + m;
        float v = acc[i][j][r];
        if (bias) v += bias[col];
        Out[(size_t)row * N + col] = v;
      }
}

// ---------------------------------------------------------------------------
// Kernel 2: per-head RMSNorm + RoPE, in place on qkv workspace.
// ---------------------------------------------------------------------------
__global__ __launch_bounds__(128) void norm_rope_kernel(
    float* __restrict__ qkv, const float* __restrict__ qw,
    const float* __restrict__ kw, const float* __restrict__ cosb,
    const float* __restrict__ sinb) {
  const int id = blockIdx.x;
  const int which = id & 1;
  const int t = id >> 1;
  const int h = t & 15;
  const int bs = t >> 4;
  const size_t base = (size_t)bs * 6144 + which * 2048 + h * 128;
  const int d = threadIdx.x;

  __shared__ float red[128];
  __shared__ float sh[128];

  const float v = qkv[base + d];
  red[d] = v * v;
  __syncthreads();
#pragma unroll
  for (int s = 64; s > 0; s >>= 1) {
    if (d < s) red[d] += red[d + s];
    __syncthreads();
  }
  const float inv = rsqrtf(red[0] * (1.0f / 128.0f) + 1e-6f);
  const float* w = which ? kw : qw;
  sh[d] = v * inv * w[d];
  __syncthreads();

  const int i = d >> 1;
  const float c = cosb[(size_t)bs * 64 + i];
  const float s = sinb[(size_t)bs * 64 + i];
  const float xr = sh[2 * i], xi = sh[2 * i + 1];
  qkv[base + d] = (d & 1) ? (xr * s + xi * c) : (xr * c - xi * s);
}

// ---------------------------------------------------------------------------
// Kernel 3: flash attention per (b, h, 64 q-rows). 128 threads = 4 waves.
// ---------------------------------------------------------------------------
#define DSTRIDE 132
#define PSTRIDE 20

__global__ __launch_bounds__(128) void attn_kernel(
    const float* __restrict__ qkv, float* __restrict__ o) {
  __shared__ float qlds[64 * DSTRIDE];
  __shared__ float klds[16 * DSTRIDE];
  __shared__ float vlds[16 * DSTRIDE];
  __shared__ float plds[4 * 16 * PSTRIDE];

  const int tid  = threadIdx.x;
  const int wave = tid >> 5, lane = tid & 31;
  const int half = lane >> 4, m = lane & 15;
  const int blk = blockIdx.x;
  const int qt = blk & 15;
  const int bh = blk >> 4;
  const int h = bh & 15;
  const int b = bh >> 4;
  const int q0 = qt * 64;
  const float scale = 0.08838834764831845f;  // 1/sqrt(128) folded into q

#pragma unroll
  for (int i = 0; i < 16; ++i) {
    int slot = tid + i * 128;
    int r = slot >> 5, c4 = slot & 31;
    float4 v = *(const float4*)(qkv + (size_t)(b * 1024 + q0 + r) * 6144 +
                                h * 128 + c4 * 4);
    v.x *= scale; v.y *= scale; v.z *= scale; v.w *= scale;
    *(float4*)(qlds + r * DSTRIDE + c4 * 4) = v;
  }

  float mi[8], li[8];
  v8f acc[8];
#pragma unroll
  for (int r = 0; r < 8; ++r) { mi[r] = -3.0e38f; li[r] = 0.0f; }
#pragma unroll
  for (int n = 0; n < 8; ++n)
#pragma unroll
    for (int r = 0; r < 8; ++r) acc[n][r] = 0.0f;

  for (int kt = 0; kt < 64; ++kt) {
    __syncthreads();
    // ---- stage k/v tiles (16x128 f32 each) ----
#if USE_ASYNC_LDS
#pragma unroll
    for (int i = 0; i < 4; ++i) {
      int slot = tid + i * 128;
      int r = slot >> 5, c4 = slot & 31;
      size_t g = (size_t)(b * 1024 + kt * 16 + r) * 6144 + h * 128 + c4 * 4;
      __builtin_amdgcn_global_load_async_to_lds_b128(
          (gv4i*)(qkv + g + 2048), (lv4i*)(klds + r * DSTRIDE + c4 * 4), 0, 0);
      __builtin_amdgcn_global_load_async_to_lds_b128(
          (gv4i*)(qkv + g + 4096), (lv4i*)(vlds + r * DSTRIDE + c4 * 4), 0, 0);
    }
#if __has_builtin(__builtin_amdgcn_s_wait_asynccnt)
    __builtin_amdgcn_s_wait_asynccnt(0);
#else
    asm volatile("s_wait_asynccnt 0" ::: "memory");
#endif
#else
#pragma unroll
    for (int i = 0; i < 4; ++i) {
      int slot = tid + i * 128;
      int r = slot >> 5, c4 = slot & 31;
      size_t g = (size_t)(b * 1024 + kt * 16 + r) * 6144 + h * 128 + c4 * 4;
      *(float4*)(klds + r * DSTRIDE + c4 * 4) = *(const float4*)(qkv + g + 2048);
      *(float4*)(vlds + r * DSTRIDE + c4 * 4) = *(const float4*)(qkv + g + 4096);
    }
#endif
    __syncthreads();

    // ---- scores (16x16) = q(16x128) @ k^T(128x16), fp32 WMMA ----
    v8f c;
#pragma unroll
    for (int r = 0; r < 8; ++r) c[r] = 0.0f;
#pragma unroll
    for (int kk = 0; kk < 32; ++kk) {
      const int ko = kk * 4 + 2 * half;
      v2f a, bb;
      a.x  = qlds[(wave * 16 + m) * DSTRIDE + ko];
      a.y  = qlds[(wave * 16 + m) * DSTRIDE + ko + 1];
      bb.x = klds[m * DSTRIDE + ko];
      bb.y = klds[m * DSTRIDE + ko + 1];
      c = wmma_f32(a, bb, c);
    }

    // ---- online softmax (rows r+8*half, 16-lane groups) ----
    float alpha[8];
#pragma unroll
    for (int r = 0; r < 8; ++r) {
      float rm = c[r];
#pragma unroll
      for (int off = 1; off < 16; off <<= 1)
        rm = fmaxf(rm, __shfl_xor(rm, off, 32));
      const float nm = fmaxf(mi[r], rm);
      alpha[r] = __expf(mi[r] - nm);
      mi[r] = nm;
      const float p = __expf(c[r] - nm);
      float rs = p;
#pragma unroll
      for (int off = 1; off < 16; off <<= 1) rs += __shfl_xor(rs, off, 32);
      li[r] = li[r] * alpha[r] + rs;
      plds[wave * (16 * PSTRIDE) + (r + 8 * half) * PSTRIDE + m] = p;
    }
#pragma unroll
    for (int n = 0; n < 8; ++n)
#pragma unroll
      for (int r = 0; r < 8; ++r) acc[n][r] *= alpha[r];

    // ---- acc += P(16x16) @ V(16x128) ----
#pragma unroll
    for (int kk = 0; kk < 4; ++kk) {
      const int ko = kk * 4 + 2 * half;
      v2f a;
      a.x = plds[wave * (16 * PSTRIDE) + m * PSTRIDE + ko];
      a.y = plds[wave * (16 * PSTRIDE) + m * PSTRIDE + ko + 1];
#pragma unroll
      for (int n = 0; n < 8; ++n) {
        v2f bb;
        bb.x = vlds[ko * DSTRIDE + n * 16 + m];
        bb.y = vlds[(ko + 1) * DSTRIDE + n * 16 + m];
        acc[n] = wmma_f32(a, bb, acc[n]);
      }
    }
  }

#pragma unroll
  for (int r = 0; r < 8; ++r) {
    const float inv = 1.0f / li[r];
    const int row = q0 + wave * 16 + r + 8 * half;
    const size_t ob = (size_t)(b * 1024 + row) * 2048 + h * 128;
#pragma unroll
    for (int n = 0; n < 8; ++n) o[ob + n * 16 + m] = acc[n][r] * inv;
  }
}

// ---------------------------------------------------------------------------
// Launch pipeline.
// ---------------------------------------------------------------------------
extern "C" void kernel_launch(void* const* d_in, const int* in_sizes, int n_in,
                              void* d_out, int out_size, void* d_ws,
                              size_t ws_size, hipStream_t stream) {
  (void)in_sizes; (void)n_in; (void)out_size; (void)ws_size;
  const float* x     = (const float*)d_in[0];
  const float* Wqkv  = (const float*)d_in[1];
  const float* qw    = (const float*)d_in[2];
  const float* kw    = (const float*)d_in[3];
  const float* Wproj = (const float*)d_in[4];
  const float* bproj = (const float*)d_in[5];
  const float* fcos  = (const float*)d_in[6];
  const float* fsin  = (const float*)d_in[7];
  float* out = (float*)d_out;

  float* qkv_ws  = (float*)d_ws;                  // 4096 x 6144 f32
  float* attn_ws = qkv_ws + (size_t)4096 * 6144;  // 4096 x 2048 f32

  gemm_bf16x3_kernel<<<dim3(48, 64), 256, 0, stream>>>(x, Wqkv, nullptr,
                                                       qkv_ws, 4096, 6144, 2048);
  norm_rope_kernel<<<dim3(4 * 1024 * 16 * 2), 128, 0, stream>>>(
      qkv_ws, qw, kw, fcos, fsin);
  attn_kernel<<<dim3(1024), 128, 0, stream>>>(qkv_ws, attn_ws);
  gemm_bf16x3_kernel<<<dim3(16, 64), 256, 0, stream>>>(attn_ws, Wproj, bproj,
                                                       out, 4096, 2048, 2048);
}
